// LearnableSinkhornLayer_28810640621614
// MI455X (gfx1250) — matrix operand order repaired and checked by reading, not compiled, not run on hardware
//
#include <hip/hip_runtime.h>
#include <math.h>

typedef _Float16 h16;
typedef __attribute__((ext_vector_type(16))) _Float16 v16h;
typedef __attribute__((ext_vector_type(8)))  float    v8f;
typedef __attribute__((ext_vector_type(4)))  unsigned u32x4;
typedef __attribute__((ext_vector_type(4)))  float    f32x4;

#define EPSV 1e-9f
#define DUST_SCALE 0.3f
#define SINK_ITERS 100
#define FD 128
#define NEG_INF (-__builtin_huge_valf())

struct Params { float temp; float dust; unsigned smax_bits; };

// ---------------- logsumexp helpers (1 exp per element) ----------------
__device__ __forceinline__ void lse_add(float& m, float& s, float x) {
  float d = x - m;                   // m may be -inf only before any data; then d=+inf
  float e = __expf(fminf(d, -d));    // exp(-|d|); exp(-inf)=0
  if (d > 0.0f) { s = s * e + 1.0f; m = x; } else { s += e; }
}
__device__ __forceinline__ void lse_merge(float& m, float& s, float mo, float so) {
  float M = fmaxf(m, mo);
  float a = (m  == M) ? s  : s  * __expf(m  - M);
  float b = (mo == M) ? so : so * __expf(mo - M);
  m = M; s = a + b;
}

// ---------------- init: params, log_mu/log_nu, u=v=0 ----------------
__global__ void k_init(Params* __restrict__ p, float* __restrict__ log_mu,
                       float* __restrict__ log_nu, float* __restrict__ u,
                       float* __restrict__ v, const float* __restrict__ lt,
                       const float* __restrict__ dsc, const int* __restrict__ ovp,
                       int N, int M) {
  const int ov = *ovp;
  const float dust = *dsc;
  int uniq_src = N - ov; if (uniq_src < 0) uniq_src = 0;
  int uniq_tgt = M - ov; if (uniq_tgt < 0) uniq_tgt = 0;
  const float mu_dust = fmaxf((float)uniq_tgt * DUST_SCALE, EPSV);
  const float nu_dust = fmaxf((float)uniq_src * DUST_SCALE, EPSV);
  const int mu_ones = (uniq_src > 0) ? ov : N;
  const int nu_ones = (uniq_tgt > 0) ? ov : M;
  const float musum = (float)mu_ones + (float)(N - mu_ones) * EPSV + mu_dust;
  const float nusum = (float)nu_ones + (float)(M - nu_ones) * EPSV + nu_dust;
  const float lms = logf(musum), lns = logf(nusum);
  const float leps = logf(EPSV);

  if (threadIdx.x == 0) {
    p->temp = fmaxf(expf(*lt), EPSV);
    p->dust = dust;
    p->smax_bits = 0xFF800000u;   // bits of -inf; atomicMin over score bits == float max (scores <= -0)
  }
  for (int i = threadIdx.x; i <= N; i += blockDim.x) {
    float lm = (i == N) ? logf(mu_dust) : ((uniq_src > 0 && i >= ov) ? leps : 0.0f);
    log_mu[i] = lm - lms;
    u[i] = 0.0f;
  }
  for (int j = threadIdx.x; j <= M; j += blockDim.x) {
    float ln = (j == M) ? logf(nu_dust) : ((uniq_tgt > 0 && j >= ov) ? leps : 0.0f);
    log_nu[j] = ln - lns;
    v[j] = 0.0f;
  }
}

// ---------------- tiny MLP: y[p][o] = act(x[p]@W + b) ----------------
__global__ void k_linear(const float* __restrict__ x, const float* __restrict__ W,
                         const float* __restrict__ b, float* __restrict__ y,
                         int P, int K, int O, int relu) {
  int idx = blockIdx.x * blockDim.x + threadIdx.x;
  if (idx >= P * O) return;
  int pp = idx / O, o = idx - pp * O;
  float acc = b[o];
  for (int k = 0; k < K; ++k) acc = fmaf(x[pp * K + k], W[k * O + o], acc);
  if (relu) acc = fmaxf(acc, 0.0f);
  y[idx] = acc;
}

// f32 feature -> (hi, lo) f16 split for precision-recovered WMMA
__global__ void k_split(const float* __restrict__ f, h16* __restrict__ hi,
                        h16* __restrict__ lo, int n) {
  int i = blockIdx.x * blockDim.x + threadIdx.x;
  if (i >= n) return;
  float x = f[i];
  h16 h = (h16)x;
  hi[i] = h;
  lo[i] = (h16)(x - (float)h);
}

__global__ void k_norm(const float* __restrict__ f, float* __restrict__ nrm, int P) {
  int p = blockIdx.x * blockDim.x + threadIdx.x;
  if (p >= P) return;
  float s = 0.0f;
  for (int c = 0; c < FD; ++c) { float x = f[(size_t)p * FD + c]; s = fmaf(x, x, s); }
  nrm[p] = s;
}

// ---------------- WMMA score GEMM: S[i][j] = -dist(i,j)/temp (f16), global max via bit atomicMin ----
__global__ __launch_bounds__(256) void k_scores(
    const h16* __restrict__ Ahh, const h16* __restrict__ Alo,
    const h16* __restrict__ Bhh, const h16* __restrict__ Blo,
    const float* __restrict__ nA, const float* __restrict__ nB,
    h16* __restrict__ S, Params* __restrict__ prm, int M) {
  const int lane = threadIdx.x & 31;
  const int wv   = threadIdx.x >> 5;
  const int wm = wv & 3, wn = wv >> 2;               // 4x2 waves -> 128x128 block tile
  const int rowBase = blockIdx.y * 128 + wm * 32;
  const int colBase = blockIdx.x * 128 + wn * 64;
  const int fr = lane & 15;
  const int kA = (lane >> 4) << 3;                   // A: K {0-7,16-23} / {8-15,24-31}
  const int kB = (lane >> 4) << 3;                   // B: K {0-7,16-23} / {8-15,24-31}

  v8f zf = {};
  v8f acc[2][4];
#pragma unroll
  for (int i = 0; i < 2; ++i)
#pragma unroll
    for (int j = 0; j < 4; ++j) acc[i][j] = zf;

  for (int k0 = 0; k0 < FD; k0 += 32) {
    v16h ah[2], al[2], bh[4], bl[4];
#pragma unroll
    for (int tm = 0; tm < 2; ++tm) {
      size_t o = (size_t)(rowBase + tm * 16 + fr) * FD + k0 + kA;
      ((u32x4*)&ah[tm])[0] = *(const u32x4*)(Ahh + o);
      ((u32x4*)&ah[tm])[1] = *(const u32x4*)(Ahh + o + 16);
      ((u32x4*)&al[tm])[0] = *(const u32x4*)(Alo + o);
      ((u32x4*)&al[tm])[1] = *(const u32x4*)(Alo + o + 16);
    }
#pragma unroll
    for (int tn = 0; tn < 4; ++tn) {
      size_t o = (size_t)(colBase + tn * 16 + fr) * FD + k0 + kB;
      ((u32x4*)&bh[tn])[0] = *(const u32x4*)(Bhh + o);
      ((u32x4*)&bh[tn])[1] = *(const u32x4*)(Bhh + o + 16);
      ((u32x4*)&bl[tn])[0] = *(const u32x4*)(Blo + o);
      ((u32x4*)&bl[tn])[1] = *(const u32x4*)(Blo + o + 16);
    }
#pragma unroll
    for (int tm = 0; tm < 2; ++tm)
#pragma unroll
      for (int tn = 0; tn < 4; ++tn) {
        acc[tm][tn] = __builtin_amdgcn_wmma_f32_16x16x32_f16(false, ah[tm], false, bl[tn], (short)0, acc[tm][tn], false, false);
        acc[tm][tn] = __builtin_amdgcn_wmma_f32_16x16x32_f16(false, al[tm], false, bh[tn], (short)0, acc[tm][tn], false, false);
        acc[tm][tn] = __builtin_amdgcn_wmma_f32_16x16x32_f16(false, ah[tm], false, bh[tn], (short)0, acc[tm][tn], false, false);
      }
  }

  const float invt = 1.0f / prm->temp;
  const int cN = lane & 15;
  const int rOff = (lane >> 4) << 3;
  float na[2][8];
#pragma unroll
  for (int tm = 0; tm < 2; ++tm)
#pragma unroll
    for (int r = 0; r < 8; ++r) na[tm][r] = nA[rowBase + tm * 16 + rOff + r];

  float smax = NEG_INF;
#pragma unroll
  for (int tm = 0; tm < 2; ++tm)
#pragma unroll
    for (int tn = 0; tn < 4; ++tn) {
      int col = colBase + tn * 16 + cN;
      float nb = nB[col];
#pragma unroll
      for (int r = 0; r < 8; ++r) {
        int row = rowBase + tm * 16 + rOff + r;
        float d2 = fmaxf(na[tm][r] + nb - 2.0f * acc[tm][tn][r], 0.0f);
        float sc = -sqrtf(d2) * invt;
        smax = fmaxf(smax, sc);
        S[(size_t)row * M + col] = (h16)fmaxf(sc, -60000.0f);   // clamp keeps f16 finite
      }
    }
  for (int d = 16; d; d >>= 1) smax = fmaxf(smax, __shfl_xor(smax, d));
  if (lane == 0) atomicMin(&prm->smax_bits, __float_as_uint(smax));
}

// ---------------- Sinkhorn row pass: u[i] = log_mu[i] - lse_j(E[i][:] + v) ----------------
__global__ __launch_bounds__(512) void k_row(
    const h16* __restrict__ S, const float* __restrict__ v,
    const float* __restrict__ log_mu, float* __restrict__ u,
    const Params* __restrict__ p, int N, int M) {
  __shared__ float redM[16], redS[16];
  const int tid = threadIdx.x, lane = tid & 31, wv = tid >> 5;
  const float smax = __uint_as_float(p->smax_bits);
  const float dust = p->dust;

  if ((int)blockIdx.x < N / 32) {
    const int c0 = tid * 16;                  // 512 threads x 16 cols = 8192
    float vv[16];
#pragma unroll
    for (int c = 0; c < 16; ++c) vv[c] = v[c0 + c] - smax;
    const float vdust = dust + v[M];
    const int row0 = blockIdx.x * 32;
    for (int r = 0; r < 32; ++r) {
      const int row = row0 + r;
      union { u32x4 q[2]; h16 h[16]; } ld;
      const u32x4* sp = (const u32x4*)(S + (size_t)row * M + c0);
      ld.q[0] = sp[0]; ld.q[1] = sp[1];
      float m = (float)ld.h[0] + vv[0];
      float s = 1.0f;
#pragma unroll
      for (int c = 1; c < 16; ++c) lse_add(m, s, (float)ld.h[c] + vv[c]);
      for (int d = 16; d; d >>= 1) { float mo = __shfl_xor(m, d), so = __shfl_xor(s, d); lse_merge(m, s, mo, so); }
      if (lane == 0) { redM[wv] = m; redS[wv] = s; }
      __syncthreads();
      if (wv == 0) {
        float mm = NEG_INF, ss = 0.0f;
        if (lane < 16) { mm = redM[lane]; ss = redS[lane]; }
        for (int d = 16; d; d >>= 1) { float mo = __shfl_xor(mm, d), so = __shfl_xor(ss, d); lse_merge(mm, ss, mo, so); }
        if (lane == 0) {
          lse_merge(mm, ss, vdust, 1.0f);                   // dustbin column
          u[row] = log_mu[row] - (mm + __logf(ss));
        }
      }
      __syncthreads();
    }
  } else {
    // dustbin row: u[N] = log_mu[N] - (dust + lse(v[0..M]))
    float m = NEG_INF, s = 0.0f;
    for (int j = tid; j <= M; j += 512) lse_add(m, s, v[j]);
    for (int d = 16; d; d >>= 1) { float mo = __shfl_xor(m, d), so = __shfl_xor(s, d); lse_merge(m, s, mo, so); }
    if (lane == 0) { redM[wv] = m; redS[wv] = s; }
    __syncthreads();
    if (wv == 0) {
      float mm = NEG_INF, ss = 0.0f;
      if (lane < 16) { mm = redM[lane]; ss = redS[lane]; }
      for (int d = 16; d; d >>= 1) { float mo = __shfl_xor(mm, d), so = __shfl_xor(ss, d); lse_merge(mm, ss, mo, so); }
      if (lane == 0) u[N] = log_mu[N] - (dust + mm + __logf(ss));
    }
  }
}

// ---------------- Sinkhorn col pass: split-K partials over 64 row chunks ----------------
__global__ __launch_bounds__(256) void k_colpart(
    const h16* __restrict__ S, const float* __restrict__ u,
    float* __restrict__ pM, float* __restrict__ pS,
    const Params* __restrict__ p, int N, int M, int CC) {
  const int cb = blockIdx.x % CC;
  const int rb = blockIdx.x / CC;
  const int c0 = cb * 2048 + threadIdx.x * 8;
  const int rpc = N >> 6;                    // rows per chunk (128)
  const int r0 = rb * rpc;
  const float smax = __uint_as_float(p->smax_bits);
  float m[8], s[8];
  {
    float a = u[r0] - smax;
    union { u32x4 q; h16 h[8]; } ld;
    ld.q = *(const u32x4*)(S + (size_t)r0 * M + c0);
#pragma unroll
    for (int c = 0; c < 8; ++c) { m[c] = (float)ld.h[c] + a; s[c] = 1.0f; }
  }
  for (int r = 1; r < rpc; ++r) {
    float a = u[r0 + r] - smax;
    union { u32x4 q; h16 h[8]; } ld;
    ld.q = *(const u32x4*)(S + (size_t)(r0 + r) * M + c0);
#pragma unroll
    for (int c = 0; c < 8; ++c) lse_add(m[c], s[c], (float)ld.h[c] + a);
  }
#pragma unroll
  for (int c = 0; c < 8; ++c) {
    pM[(size_t)rb * M + c0 + c] = m[c];
    pS[(size_t)rb * M + c0 + c] = s[c];
  }
}

__global__ __launch_bounds__(256) void k_colfin(
    const float* __restrict__ pM, const float* __restrict__ pS,
    const float* __restrict__ u, const float* __restrict__ log_nu,
    float* __restrict__ v, const Params* __restrict__ p, int N, int M) {
  __shared__ float redM[16], redS[16];
  const float dust = p->dust;
  if ((int)blockIdx.x < M / 256) {
    int j = blockIdx.x * 256 + threadIdx.x;
    float m = pM[j], s = pS[j];
    for (int k = 1; k < 64; ++k) lse_merge(m, s, pM[(size_t)k * M + j], pS[(size_t)k * M + j]);
    lse_merge(m, s, dust + u[N], 1.0f);                 // dustbin row
    v[j] = log_nu[j] - (m + __logf(s));
  } else {
    // dustbin col: v[M] = log_nu[M] - (dust + lse(u[0..N]))
    const int tid = threadIdx.x, lane = tid & 31, wv = tid >> 5;
    float m = NEG_INF, s = 0.0f;
    for (int i = tid; i <= N; i += 256) lse_add(m, s, u[i]);
    for (int d = 16; d; d >>= 1) { float mo = __shfl_xor(m, d), so = __shfl_xor(s, d); lse_merge(m, s, mo, so); }
    if (lane == 0) { redM[wv] = m; redS[wv] = s; }
    __syncthreads();
    if (wv == 0) {
      float mm = NEG_INF, ss = 0.0f;
      if (lane < 8) { mm = redM[lane]; ss = redS[lane]; }
      for (int d = 16; d; d >>= 1) { float mo = __shfl_xor(mm, d), so = __shfl_xor(ss, d); lse_merge(mm, ss, mo, so); }
      if (lane == 0) v[M] = log_nu[M] - (dust + mm + __logf(ss));
    }
  }
}

// ---------------- outputs ----------------
__global__ __launch_bounds__(256) void k_transport(
    const h16* __restrict__ S, const float* __restrict__ u,
    const float* __restrict__ v, float* __restrict__ out,
    const Params* __restrict__ p, int N, int M, int CC) {
  const int row = blockIdx.x / CC;
  const int c0 = (blockIdx.x % CC) * 2048 + threadIdx.x * 8;
  const float a = u[row] - __uint_as_float(p->smax_bits);
  union { u32x4 q; h16 h[8]; } ld;
  ld.q = *(const u32x4*)(S + (size_t)row * M + c0);
  f32x4 o0, o1;
#pragma unroll
  for (int c = 0; c < 4; ++c) o0[c] = __expf((float)ld.h[c] + a + v[c0 + c]);
#pragma unroll
  for (int c = 0; c < 4; ++c) o1[c] = __expf((float)ld.h[4 + c] + a + v[c0 + 4 + c]);
  float* dst = out + (size_t)row * M + c0;
  *(f32x4*)dst = o0;
  *(f32x4*)(dst + 4) = o1;
}

__global__ void k_edges(const float* __restrict__ u, const float* __restrict__ v,
                        float* __restrict__ out, const Params* __restrict__ p,
                        int N, int M) {
  int i = blockIdx.x * blockDim.x + threadIdx.x;
  const float dust = p->dust;
  const size_t NM = (size_t)N * M;
  if (i < N) out[NM + i] = __expf(dust + u[i] + v[M]);            // transport[:-1, -1]
  int j = i - N;
  if (j >= 0 && j < M) out[NM + N + j] = __expf(dust + u[N] + v[j]); // transport[-1, :-1]
}

// ---------------- host orchestration ----------------
extern "C" void kernel_launch(void* const* d_in, const int* in_sizes, int n_in,
                              void* d_out, int out_size, void* d_ws, size_t ws_size,
                              hipStream_t stream) {
  const int N = in_sizes[0] / 3;
  const int M = in_sizes[1] / 3;
  const float* src = (const float*)d_in[0];
  const float* tgt = (const float*)d_in[1];
  const float* sw1 = (const float*)d_in[2];  const float* sb1 = (const float*)d_in[3];
  const float* sw2 = (const float*)d_in[4];  const float* sb2 = (const float*)d_in[5];
  const float* sw3 = (const float*)d_in[6];  const float* sb3 = (const float*)d_in[7];
  const float* tw1 = (const float*)d_in[8];  const float* tb1 = (const float*)d_in[9];
  const float* tw2 = (const float*)d_in[10]; const float* tb2 = (const float*)d_in[11];
  const float* tw3 = (const float*)d_in[12]; const float* tb3 = (const float*)d_in[13];
  const float* lt  = (const float*)d_in[14];
  const float* dsc = (const float*)d_in[15];
  const int*   ovp = (const int*)d_in[16];
  (void)n_in; (void)out_size; (void)ws_size;

  char* ws = (char*)d_ws;
  size_t off = 0;
  auto take = [&](size_t bytes) -> char* {
    char* pp = ws + off;
    off = (off + bytes + 255) & ~(size_t)255;
    return pp;
  };

  h16*    S      = (h16*)   take((size_t)N * M * 2);   // 134 MB: fits MI455X 192 MB L2
  Params* prm    = (Params*)take(256);
  float*  log_mu = (float*) take((size_t)(N + 1) * 4);
  float*  log_nu = (float*) take((size_t)(M + 1) * 4);
  float*  u      = (float*) take((size_t)(N + 1) * 4);
  float*  v      = (float*) take((size_t)(M + 1) * 4);
  float*  nA     = (float*) take((size_t)N * 4);
  float*  nB     = (float*) take((size_t)M * 4);
  float*  fA     = (float*) take((size_t)N * FD * 4);
  float*  fB     = (float*) take((size_t)M * FD * 4);
  const int P = (N > M) ? N : M;
  float*  h1     = (float*) take((size_t)P * 64 * 4);
  float*  h2     = (float*) take((size_t)P * FD * 4);
  h16*    Ahh    = (h16*)   take((size_t)N * FD * 2);
  h16*    Alo    = (h16*)   take((size_t)N * FD * 2);
  h16*    Bhh    = (h16*)   take((size_t)M * FD * 2);
  h16*    Blo    = (h16*)   take((size_t)M * FD * 2);
  float*  pM     = (float*) take((size_t)64 * M * 4);
  float*  pS     = (float*) take((size_t)64 * M * 4);

  k_init<<<1, 256, 0, stream>>>(prm, log_mu, log_nu, u, v, lt, dsc, ovp, N, M);

  // MLPs (src -> fA, tgt -> fB)
  k_linear<<<(N * 64  + 255) / 256, 256, 0, stream>>>(src, sw1, sb1, h1, N, 3, 64, 1);
  k_linear<<<(N * 128 + 255) / 256, 256, 0, stream>>>(h1, sw2, sb2, h2, N, 64, 128, 1);
  k_linear<<<(N * 128 + 255) / 256, 256, 0, stream>>>(h2, sw3, sb3, fA, N, 128, 128, 0);
  k_linear<<<(M * 64  + 255) / 256, 256, 0, stream>>>(tgt, tw1, tb1, h1, M, 3, 64, 1);
  k_linear<<<(M * 128 + 255) / 256, 256, 0, stream>>>(h1, tw2, tb2, h2, M, 64, 128, 1);
  k_linear<<<(M * 128 + 255) / 256, 256, 0, stream>>>(h2, tw3, tb3, fB, M, 128, 128, 0);

  k_split<<<(N * FD + 255) / 256, 256, 0, stream>>>(fA, Ahh, Alo, N * FD);
  k_split<<<(M * FD + 255) / 256, 256, 0, stream>>>(fB, Bhh, Blo, M * FD);
  k_norm<<<(N + 255) / 256, 256, 0, stream>>>(fA, nA, N);
  k_norm<<<(M + 255) / 256, 256, 0, stream>>>(fB, nB, M);

  // WMMA score matrix + global max
  k_scores<<<dim3(M / 128, N / 128), 256, 0, stream>>>(Ahh, Alo, Bhh, Blo, nA, nB, S, prm, M);

  // Sinkhorn iterations
  const int CC = M / 2048;
  for (int it = 0; it < SINK_ITERS; ++it) {
    k_row<<<N / 32 + 1, 512, 0, stream>>>(S, v, log_mu, u, prm, N, M);
    k_colpart<<<CC * 64, 256, 0, stream>>>(S, u, pM, pS, prm, N, M, CC);
    k_colfin<<<M / 256 + 1, 256, 0, stream>>>(pM, pS, u, log_nu, v, prm, N, M);
  }

  float* out = (float*)d_out;
  k_transport<<<N * CC, 256, 0, stream>>>(S, u, v, out, prm, N, M, CC);
  k_edges<<<(N + M + 255) / 256, 256, 0, stream>>>(u, v, out, prm, N, M);
}
